// SEGNNModel_36094905155705
// MI455X (gfx1250) — matrix operand order repaired and checked by reading, not compiled, not run on hardware
//
#include <hip/hip_runtime.h>
#include <math.h>

// ---------------- problem constants ----------------
#define F_INC   9
#define HC      256
#define NSC     130
#define NVC     42
#define NGC     172          // NS + NV
#define NNODES  10000
#define NEDGES  160000
#define AGGC    260          // 130 scalars + 126 vec + 1 sh_s + 3 sh_v
#define SQRT3C     1.7320508075688772f
#define INV_SQRT3  0.5773502691896258f

#define MODE_STAGE   0
#define MODE_SCATTER 1
#define MODE_FINAL   2

typedef unsigned short u16;
typedef __attribute__((ext_vector_type(16))) __bf16 v16bf;
typedef __attribute__((ext_vector_type(8)))  float  v8f;

// ---------------- small helpers ----------------
__device__ __forceinline__ u16 f2bf(float f) {
    unsigned u = __builtin_bit_cast(unsigned, f);
    u += 0x7FFFu + ((u >> 16) & 1u);           // round-to-nearest-even
    return (u16)(u >> 16);
}
__device__ __forceinline__ v8f zero8() {
    v8f z = {0.f,0.f,0.f,0.f,0.f,0.f,0.f,0.f};
    return z;
}
__device__ __forceinline__ float sigm(float x) { return 1.f / (1.f + __expf(-x)); }

// ---- fragment-major A storage in LDS ----
// WMMA 16-bit A layout: lane(m = lane&15, hi = lane>>4): elems 0-7 = K kt*32+hi*8+(0..7),
// elems 8-15 = K kt*32+hi*8+16+(0..7).  We store each lane's 16 elems contiguously:
//    index = (kt*32 + hi*16 + m)*16 + elem
__device__ __forceinline__ void sta(u16* __restrict__ A, int r, int k, u16 v) {
    const int kt   = k >> 5;
    const int kk   = k & 31;
    const int hi   = (kk >> 3) & 1;
    const int elem = (kk & 7) | (((kk >> 4) & 1) << 3);
    A[((kt << 5) + (hi << 4) + (r & 15)) * 16 + elem] = v;
}
__device__ __forceinline__ v16bf load_a_frag(const u16* __restrict__ A, int kt, int lane) {
    return *reinterpret_cast<const v16bf*>(A + (((size_t)(kt << 5) + lane) << 4));
}
// ---- fragment-major B storage in global ws ----
// One contiguous 32B chunk per lane per (kt,nt) tile.
__device__ __forceinline__ v16bf load_b_frag(const u16* __restrict__ W, int NT, int kt, int nt, int lane) {
    return *reinterpret_cast<const v16bf*>(W + (((size_t)((kt * NT + nt) << 5) + lane) << 4));
}

__device__ __forceinline__ v8f wmma_bf16(v16bf a, v16bf b, v8f c) {
    return __builtin_amdgcn_wmma_f32_16x16x32_bf16(false, a, false, b, (short)0, c, false, false);
}

// ---------------- tensor-product descriptor ----------------
struct TP {
    const u16*  Wss;   // fragment-major (KsP/32) x (NpS/16) tiles
    const u16*  Wvv;
    const u16*  Wsv;
    const u16*  Wvs;
    const float* bias; // NSo fp32
    int  KsP, KvP, NpS, NpV, NSo, NVo, NSsc;
    float scale;
    int  gated;
};

// One wave computes a full 16-row tensor-product (+gate) stage.
//  out_s = scale*(s2*(As@Wss) + Avd@Wvv) + b      (swish / sigmoid-gate split if gated)
//  out_v[o][m] = g * scale*((As@Wsv)[o]*v2[m] + s2*(Avm_m@Wvs)[o])
__device__ void tp_wave(const TP t,
                        const u16* __restrict__ As,
                        const u16* __restrict__ Avd,
                        const u16* __restrict__ Avm,
                        const float* __restrict__ s2,
                        const float* __restrict__ v2,
                        float* __restrict__ gateBuf,
                        int mode,
                        u16* oAs, u16* oAvd, u16* oAvm,
                        float* agg, const int* rowId,
                        float* xs_out, float* xv_out,
                        const float* xs_in, const float* xv_in, int rec)
{
    const int lane = threadIdx.x & 31;
    const int nhi  = (lane >> 4) << 3;   // row base (0 or 8)
    const int nlo  = lane & 15;          // column within tile
    const int nkt_s = t.KsP >> 5;
    const int nkt_v = t.KvP >> 5;
    const int NTs   = t.NpS >> 4;
    const int NTv   = t.NpV >> 4;

    // -------- scalar-output tiles --------
    for (int nt = 0; nt < NTs; ++nt) {
        v8f acc_ss = zero8();
        for (int kt = 0; kt < nkt_s; ++kt) {
            v16bf a = load_a_frag(As, kt, lane);
            v16bf b = load_b_frag(t.Wss, NTs, kt, nt, lane);
            acc_ss = wmma_bf16(a, b, acc_ss);
        }
        v8f acc_vv = zero8();
        for (int kt = 0; kt < nkt_v; ++kt) {
            v16bf a = load_a_frag(Avd, kt, lane);
            v16bf b = load_b_frag(t.Wvv, NTs, kt, nt, lane);
            acc_vv = wmma_bf16(a, b, acc_vv);
        }
        const int col = (nt << 4) + nlo;
#pragma unroll
        for (int i = 0; i < 8; ++i) {
            const int row = nhi + i;
            float val = t.scale * (s2[row] * acc_ss[i] + acc_vv[i]);
            if (col < t.NSo) val += t.bias[col];
            if (t.gated) {
                if (col < t.NSsc) {
                    float sw = val * sigm(val);                      // swish
                    if (mode == MODE_STAGE)
                        sta(oAs, row, col, f2bf(sw));
                    else // SCATTER
                        unsafeAtomicAdd(&agg[(size_t)rowId[row] * AGGC + col], sw);
                } else if (col < t.NSo) {
                    gateBuf[row * 48 + (col - t.NSsc)] = sigm(val);
                }
            } else if (col < t.NSo) {                                // FINAL (upd2, no gate)
                float o = val + (rec ? xs_in[(size_t)rowId[row] * 256 + col] : 0.f);
                xs_out[(size_t)rowId[row] * 256 + col] = o;
            }
        }
    }
    __syncthreads();   // gateBuf visibility for vector epilogue

    // -------- vector-output tiles --------
    for (int nt = 0; nt < NTv; ++nt) {
        v8f acc_p = zero8();
        for (int kt = 0; kt < nkt_s; ++kt) {
            v16bf a = load_a_frag(As, kt, lane);
            v16bf b = load_b_frag(t.Wsv, NTv, kt, nt, lane);
            acc_p = wmma_bf16(a, b, acc_p);
        }
        v8f acc_q0 = zero8(), acc_q1 = zero8(), acc_q2 = zero8();
        for (int kt = 0; kt < nkt_v; ++kt) {
            v16bf b  = load_b_frag(t.Wvs, NTv, kt, nt, lane);
            v16bf a0 = load_a_frag(Avm + 0 * 16 * t.KvP, kt, lane);
            acc_q0 = wmma_bf16(a0, b, acc_q0);
            v16bf a1 = load_a_frag(Avm + 1 * 16 * t.KvP, kt, lane);
            acc_q1 = wmma_bf16(a1, b, acc_q1);
            v16bf a2 = load_a_frag(Avm + 2 * 16 * t.KvP, kt, lane);
            acc_q2 = wmma_bf16(a2, b, acc_q2);
        }
        const int o = (nt << 4) + nlo;
#pragma unroll
        for (int i = 0; i < 8; ++i) {
            const int row = nhi + i;
            if (o < t.NVo) {
                float p  = acc_p[i];
                float g  = t.gated ? gateBuf[row * 48 + o] : 1.f;
                float ss = s2[row];
                float gx = g * t.scale;
                float ov0 = gx * (p * v2[row * 3 + 0] + ss * acc_q0[i]);
                float ov1 = gx * (p * v2[row * 3 + 1] + ss * acc_q1[i]);
                float ov2 = gx * (p * v2[row * 3 + 2] + ss * acc_q2[i]);
                if (mode == MODE_STAGE) {
                    float d = (ov0 * v2[row * 3 + 0] + ov1 * v2[row * 3 + 1] +
                               ov2 * v2[row * 3 + 2]) * INV_SQRT3;
                    sta(oAvd, row, o, f2bf(d));
                    sta(oAvm + 0 * 16 * 64, row, o, f2bf(ov0));
                    sta(oAvm + 1 * 16 * 64, row, o, f2bf(ov1));
                    sta(oAvm + 2 * 16 * 64, row, o, f2bf(ov2));
                } else if (mode == MODE_SCATTER) {
                    size_t b = (size_t)rowId[row] * AGGC + 130 + o * 3;
                    unsafeAtomicAdd(&agg[b + 0], ov0);
                    unsafeAtomicAdd(&agg[b + 1], ov1);
                    unsafeAtomicAdd(&agg[b + 2], ov2);
                } else { // FINAL
                    size_t b = (size_t)rowId[row] * 126 + o * 3;
                    xv_out[b + 0] = ov0 + (rec ? xv_in[b + 0] : 0.f);
                    xv_out[b + 1] = ov1 + (rec ? xv_in[b + 1] : 0.f);
                    xv_out[b + 2] = ov2 + (rec ? xv_in[b + 2] : 0.f);
                }
            }
        }
    }
    __syncthreads();
}

// ---------------- weight pack: fp32 KxN -> fragment-major zero-padded bf16 ----------------
// dst[((kt*NT + nt)*32 + lane)*16 + elem] = bf16(src[k,n]),
//   n = nt*16 + (lane&15), k = kt*32 + (lane>>4)*8 + (elem&7) + (elem>>3)*16
__global__ void k_pack(const float* __restrict__ src, u16* __restrict__ dst,
                       int K, int N, int KT, int NT) {
    int i = blockIdx.x * blockDim.x + threadIdx.x;
    int total = KT * NT * 512;
    if (i >= total) return;
    int elem = i & 15;
    int lane = (i >> 4) & 31;
    int tile = i >> 9;
    int nt = tile % NT, kt = tile / NT;
    int n = (nt << 4) + (lane & 15);
    int k = (kt << 5) + ((lane >> 4) << 3) + (elem & 7) + ((elem >> 3) << 4);
    float v = (k < K && n < N) ? src[k * N + n] : 0.f;
    dst[i] = f2bf(v);
}

__global__ void k_zero(float* __restrict__ p, int n) {
    int i = blockIdx.x * blockDim.x + threadIdx.x;
    if (i < n) p[i] = 0.f;
}

// ---------------- embedding: h = (swish(x@W1*3+b1))@W2*16+b2 ----------------
__global__ __launch_bounds__(256) void k_embed(const float* __restrict__ x,
                                               const float* __restrict__ W1,
                                               const float* __restrict__ b1,
                                               const float* __restrict__ W2,
                                               const float* __restrict__ b2,
                                               float* __restrict__ xs) {
    __shared__ float xl[F_INC];
    __shared__ float h1[HC];
    const int n = blockIdx.x, j = threadIdx.x;
    if (j < F_INC) xl[j] = x[n * F_INC + j];
    __syncthreads();
    float t = 0.f;
#pragma unroll
    for (int i = 0; i < F_INC; ++i) t += xl[i] * W1[i * HC + j];
    t = t * 3.0f + b1[j];
    h1[j] = t * sigm(t);
    __syncthreads();
    float u = 0.f;
    for (int k = 0; k < HC; ++k) u += h1[k] * W2[k * HC + j];
    xs[(size_t)n * 256 + j] = u * 16.0f + b2[j];
}

// ---------------- per-edge messages: msg1 (gate) -> msg2 (gate) -> atomic agg ----------------
__global__ __launch_bounds__(32) void k_edge(const float* __restrict__ xs,
                                             const float* __restrict__ xv,
                                             const float* __restrict__ pos,
                                             const int*   __restrict__ eidx,
                                             const float* __restrict__ cell,
                                             TP t1, TP t2,
                                             float* __restrict__ agg,
                                             float* __restrict__ deg,
                                             int S_in, int V_in) {
    __shared__ __align__(32) u16 As1[16 * 544];
    __shared__ __align__(32) u16 Avd1[16 * 96];
    __shared__ __align__(32) u16 Avm1[3 * 16 * 96];
    __shared__ __align__(32) u16 As2[16 * 160];
    __shared__ __align__(32) u16 Avd2[16 * 64];
    __shared__ __align__(32) u16 Avm2[3 * 16 * 64];
    __shared__ float s2b[16], v2b[16 * 3], distb[16], gateBuf[16 * 48];
    __shared__ int rDst[16], rSrc[16];

    const int lane = threadIdx.x;
    const int e0   = blockIdx.x * 16;

    if (lane < 16) {
        int e = e0 + lane;
        int s = eidx[e];
        int d = eidx[NEDGES + e];
        float rx = pos[d * 3 + 0] - pos[s * 3 + 0] + cell[(size_t)e * 3 + 0];
        float ry = pos[d * 3 + 1] - pos[s * 3 + 1] + cell[(size_t)e * 3 + 1];
        float rz = pos[d * 3 + 2] - pos[s * 3 + 2] + cell[(size_t)e * 3 + 2];
        float dist = rx * rx + ry * ry + rz * rz;
        float inv  = SQRT3C * rsqrtf(dist + 1e-12f);
        rDst[lane] = d; rSrc[lane] = s; distb[lane] = dist; s2b[lane] = 1.f;
        v2b[lane * 3 + 0] = rx * inv;
        v2b[lane * 3 + 1] = ry * inv;
        v2b[lane * 3 + 2] = rz * inv;
    }
    for (int i = lane; i < 16 * 160;    i += 32) As2[i]  = 0;
    for (int i = lane; i < 16 * 64;     i += 32) Avd2[i] = 0;
    for (int i = lane; i < 3 * 16 * 64; i += 32) Avm2[i] = 0;
    __syncthreads();

    // stage A_s = bf16([xs_dst | xs_src | dist | 0...]) in fragment-major order
    for (int r = 0; r < 16; ++r) {
        const float* pd = xs + (size_t)rDst[r] * 256;
        const float* ps = xs + (size_t)rSrc[r] * 256;
        float dd = distb[r];
        for (int k = lane; k < t1.KsP; k += 32) {
            float v = 0.f;
            if (k < S_in)            v = pd[k];
            else if (k < 2 * S_in)   v = ps[k - S_in];
            else if (k == 2 * S_in)  v = dd;
            sta(As1, r, k, f2bf(v));
        }
    }
    // stage vector channels (layers > 0)
    if (t1.KvP > 0) {
        for (int r = 0; r < 16; ++r) {
            float vx = v2b[r * 3 + 0], vy = v2b[r * 3 + 1], vz = v2b[r * 3 + 2];
            for (int v = lane; v < t1.KvP; v += 32) {
                float c0 = 0.f, c1 = 0.f, c2 = 0.f;
                if (v < V_in) {
                    const float* p = xv + (size_t)rDst[r] * 126 + v * 3;
                    c0 = p[0]; c1 = p[1]; c2 = p[2];
                } else if (v < 2 * V_in) {
                    const float* p = xv + (size_t)rSrc[r] * 126 + (v - V_in) * 3;
                    c0 = p[0]; c1 = p[1]; c2 = p[2];
                }
                sta(Avm1 + 0 * 16 * t1.KvP, r, v, f2bf(c0));
                sta(Avm1 + 1 * 16 * t1.KvP, r, v, f2bf(c1));
                sta(Avm1 + 2 * 16 * t1.KvP, r, v, f2bf(c2));
                sta(Avd1, r, v, f2bf((c0 * vx + c1 * vy + c2 * vz) * INV_SQRT3));
            }
        }
    }
    __syncthreads();

    tp_wave(t1, As1, Avd1, Avm1, s2b, v2b, gateBuf, MODE_STAGE,
            As2, Avd2, Avm2, nullptr, nullptr,
            nullptr, nullptr, nullptr, nullptr, 0);
    __syncthreads();
    tp_wave(t2, As2, Avd2, Avm2, s2b, v2b, gateBuf, MODE_SCATTER,
            nullptr, nullptr, nullptr, agg, rDst,
            nullptr, nullptr, nullptr, nullptr, 0);

    if (lane < 16) {
        size_t b = (size_t)rDst[lane] * AGGC;
        unsafeAtomicAdd(&agg[b + 256], 1.f);
        unsafeAtomicAdd(&agg[b + 257], v2b[lane * 3 + 0]);
        unsafeAtomicAdd(&agg[b + 258], v2b[lane * 3 + 1]);
        unsafeAtomicAdd(&agg[b + 259], v2b[lane * 3 + 2]);
        unsafeAtomicAdd(&deg[rDst[lane]], 1.f);
    }
}

// ---------------- per-node update: upd1 (gate) -> upd2 (+recurrent) ----------------
__global__ __launch_bounds__(32) void k_node(const float* __restrict__ xs_in,
                                             const float* __restrict__ xv_in,
                                             const float* __restrict__ agg,
                                             const float* __restrict__ deg,
                                             TP t1, TP t2,
                                             float* __restrict__ xs_out,
                                             float* __restrict__ xv_out,
                                             int S_in, int V_in, int rec) {
    __shared__ __align__(32) u16 As1[16 * 416];
    __shared__ __align__(32) u16 Avd1[16 * 96];
    __shared__ __align__(32) u16 Avm1[3 * 16 * 96];
    __shared__ __align__(32) u16 As2[16 * 160];
    __shared__ __align__(32) u16 Avd2[16 * 64];
    __shared__ __align__(32) u16 Avm2[3 * 16 * 64];
    __shared__ float s2b[16], v2b[16 * 3], invd[16], gateBuf[16 * 48];
    __shared__ int rId[16];

    const int lane = threadIdx.x;
    const int n0   = blockIdx.x * 16;

    if (lane < 16) {
        int n = n0 + lane;
        rId[lane] = n;
        float iv = 1.f / fmaxf(deg[n], 1.f);
        invd[lane] = iv;
        const float* ag = agg + (size_t)n * AGGC;
        s2b[lane] = ag[256] * iv;
        v2b[lane * 3 + 0] = ag[257] * iv;
        v2b[lane * 3 + 1] = ag[258] * iv;
        v2b[lane * 3 + 2] = ag[259] * iv;
    }
    for (int i = lane; i < 16 * 160;    i += 32) As2[i]  = 0;
    for (int i = lane; i < 16 * 64;     i += 32) Avd2[i] = 0;
    for (int i = lane; i < 3 * 16 * 64; i += 32) Avm2[i] = 0;
    __syncthreads();

    // A_s = bf16([xs | a_s]) in fragment-major order
    for (int r = 0; r < 16; ++r) {
        const float* px = xs_in + (size_t)rId[r] * 256;
        const float* ag = agg   + (size_t)rId[r] * AGGC;
        float iv = invd[r];
        for (int k = lane; k < t1.KsP; k += 32) {
            float v = 0.f;
            if (k < S_in)             v = px[k];
            else if (k < S_in + NSC)  v = ag[k - S_in] * iv;
            sta(As1, r, k, f2bf(v));
        }
    }
    // vector channels: [xv | a_v]
    for (int r = 0; r < 16; ++r) {
        float vx = v2b[r * 3 + 0], vy = v2b[r * 3 + 1], vz = v2b[r * 3 + 2];
        float iv = invd[r];
        const float* ag = agg + (size_t)rId[r] * AGGC;
        for (int v = lane; v < t1.KvP; v += 32) {
            float c0 = 0.f, c1 = 0.f, c2 = 0.f;
            if (v < V_in) {
                const float* p = xv_in + (size_t)rId[r] * 126 + v * 3;
                c0 = p[0]; c1 = p[1]; c2 = p[2];
            } else if (v < V_in + NVC) {
                const float* p = ag + 130 + (v - V_in) * 3;
                c0 = p[0] * iv; c1 = p[1] * iv; c2 = p[2] * iv;
            }
            sta(Avm1 + 0 * 16 * t1.KvP, r, v, f2bf(c0));
            sta(Avm1 + 1 * 16 * t1.KvP, r, v, f2bf(c1));
            sta(Avm1 + 2 * 16 * t1.KvP, r, v, f2bf(c2));
            sta(Avd1, r, v, f2bf((c0 * vx + c1 * vy + c2 * vz) * INV_SQRT3));
        }
    }
    __syncthreads();

    tp_wave(t1, As1, Avd1, Avm1, s2b, v2b, gateBuf, MODE_STAGE,
            As2, Avd2, Avm2, nullptr, nullptr,
            nullptr, nullptr, nullptr, nullptr, 0);
    __syncthreads();
    tp_wave(t2, As2, Avd2, Avm2, s2b, v2b, gateBuf, MODE_FINAL,
            nullptr, nullptr, nullptr, nullptr, rId,
            xs_out, xv_out, xs_in, xv_in, rec);
}

// ---------------- final pack: out = [xs(130) | xv(126)] ----------------
__global__ void k_out(const float* __restrict__ xs, const float* __restrict__ xv,
                      float* __restrict__ out) {
    int i = blockIdx.x * blockDim.x + threadIdx.x;
    if (i >= NNODES * 256) return;
    int n = i >> 8, c = i & 255;
    out[i] = (c < NSC) ? xs[(size_t)n * 256 + c] : xv[(size_t)n * 126 + (c - NSC)];
}

// ---------------- host ----------------
extern "C" void kernel_launch(void* const* d_in, const int* in_sizes, int n_in,
                              void* d_out, int out_size, void* d_ws, size_t ws_size,
                              hipStream_t stream) {
    (void)in_sizes; (void)n_in; (void)out_size; (void)ws_size;
    const float* x    = (const float*)d_in[0];
    const float* pos  = (const float*)d_in[1];
    const int*   eidx = (const int*)d_in[2];
    const float* cell = (const float*)d_in[3];
    // params leaves in insertion order: emb1{Wss,Wvv,Wsv,Wvs,b}, emb2{...},
    // layers[i]{msg1,msg2,upd1,upd2}x{Wss,Wvv,Wsv,Wvs,b}
    const float* emb1_Wss = (const float*)d_in[4];
    const float* emb1_b   = (const float*)d_in[8];
    const float* emb2_Wss = (const float*)d_in[9];
    const float* emb2_b   = (const float*)d_in[13];
    auto LW = [&](int layer, int tp, int leaf) -> const float* {
        return (const float*)d_in[14 + layer * 20 + tp * 5 + leaf];
    };

    char* ws = (char*)d_ws;
    size_t cur = 0;
    auto alloc = [&](size_t bytes) -> char* {
        char* p = ws + cur;
        cur += (bytes + 255) & ~(size_t)255;
        return p;
    };

    float* xsA = (float*)alloc((size_t)NNODES * 256 * 4);
    float* xsB = (float*)alloc((size_t)NNODES * 256 * 4);
    float* xvA = (float*)alloc((size_t)NNODES * 126 * 4);
    float* xvB = (float*)alloc((size_t)NNODES * 126 * 4);
    float* agg = (float*)alloc((size_t)NNODES * AGGC * 4);
    float* deg = (float*)alloc((size_t)NNODES * 4);

    auto packW = [&](const float* src, int K, int N, int Kp, int Np) -> u16* {
        int KT = Kp / 32, NT = Np / 16;
        u16* dst = (u16*)alloc((size_t)KT * NT * 512 * 2);
        k_pack<<<(KT * NT * 512 + 255) / 256, 256, 0, stream>>>(src, dst, K, N, KT, NT);
        return dst;
    };

    auto mkTP = [&](const float* Wss, const float* Wvv, const float* Wsv,
                    const float* Wvs, const float* b,
                    int Ks, int Kv, int NSo, int NVo, int gated, int NpS) -> TP {
        TP t{};
        int KsP = (Ks + 31) & ~31;
        int KvP = Kv ? ((Kv + 31) & ~31) : 0;
        t.KsP = KsP; t.KvP = KvP; t.NpS = NpS; t.NpV = 48;
        t.NSo = NSo; t.NVo = NVo;
        t.NSsc = gated ? (NSo - NVo) : NSo;
        t.gated = gated;
        t.scale = sqrtf((float)(Ks + Kv));
        t.bias  = b;
        t.Wss = packW(Wss, Ks, NSo, KsP, NpS);
        t.Wsv = packW(Wsv, Ks, NVo, KsP, 48);
        if (KvP) {
            t.Wvv = packW(Wvv, Kv, NSo, KvP, NpS);
            t.Wvs = packW(Wvs, Kv, NVo, KvP, 48);
        } else {
            t.Wvv = nullptr; t.Wvs = nullptr;
        }
        return t;
    };

    TP tmsg1[4], tmsg2[4], tupd1[4], tupd2[4];
    for (int i = 0; i < 4; ++i) {
        int S = i ? NSC : HC;
        int V = i ? NVC : 0;
        tmsg1[i] = mkTP(LW(i,0,0), LW(i,0,1), LW(i,0,2), LW(i,0,3), LW(i,0,4),
                        2 * S + 1, 2 * V, NGC, NVC, 1, 176);
        tmsg2[i] = mkTP(LW(i,1,0), LW(i,1,1), LW(i,1,2), LW(i,1,3), LW(i,1,4),
                        NSC, NVC, NGC, NVC, 1, 176);
        tupd1[i] = mkTP(LW(i,2,0), LW(i,2,1), LW(i,2,2), LW(i,2,3), LW(i,2,4),
                        S + NSC, V + NVC, NGC, NVC, 1, 176);
        tupd2[i] = mkTP(LW(i,3,0), LW(i,3,1), LW(i,3,2), LW(i,3,3), LW(i,3,4),
                        NSC, NVC, NSC, NVC, 0, 144);
    }

    k_embed<<<NNODES, 256, 0, stream>>>(x, emb1_Wss, emb1_b, emb2_Wss, emb2_b, xsA);

    float* xsIn = xsA; float* xvIn = xvA;
    float* xsOut = xsB; float* xvOut = xvB;
    for (int i = 0; i < 4; ++i) {
        int S = i ? NSC : HC;
        int V = i ? NVC : 0;
        k_zero<<<(NNODES * AGGC + 255) / 256, 256, 0, stream>>>(agg, NNODES * AGGC);
        k_zero<<<(NNODES + 255) / 256, 256, 0, stream>>>(deg, NNODES);
        k_edge<<<NEDGES / 16, 32, 0, stream>>>(xsIn, xvIn, pos, eidx, cell,
                                               tmsg1[i], tmsg2[i], agg, deg, S, V);
        k_node<<<NNODES / 16, 32, 0, stream>>>(xsIn, xvIn, agg, deg,
                                               tupd1[i], tupd2[i], xsOut, xvOut,
                                               S, V, i > 0);
        float* ts = xsIn; xsIn = xsOut; xsOut = ts;
        float* tv = xvIn; xvIn = xvOut; xvOut = tv;
    }
    k_out<<<(NNODES * 256 + 255) / 256, 256, 0, stream>>>(xsIn, xvIn, (float*)d_out);
}
